// GraphCNN_83932250898779
// MI455X (gfx1250) — compile-verified, matrix operand order checked by reading, not run on hardware
//
#include <hip/hip_runtime.h>

#define NN 100000    // nodes
#define ND 128       // node feature dim
#define ED 16        // edge feature dim
#define NE 1200000   // edges
#define HD 64        // hidden
#define NL 3         // conv layers
#define NG 1000      // graphs
#define JK (NL * HD) // 192 (JumpingKnowledge concat)
#define EKP 32       // edge-weight packed K (zero-padded 16 -> 32)

typedef __attribute__((ext_vector_type(16))) _Float16 v16h;
typedef __attribute__((ext_vector_type(8)))  _Float16 v8h;
typedef __attribute__((ext_vector_type(8)))  float    v8f;

__device__ __forceinline__ void atomic_add_f32(float* p, float v) {
  __hip_atomic_fetch_add(p, v, __ATOMIC_RELAXED, __HIP_MEMORY_SCOPE_AGENT);
}

// ---- WMMA fragment layout (CDNA5 ISA 7.12.2, 16-bit A 16x32 / B 32x16) ----
// lane L: kh = (L>>4)*8. VGPR0-3 hold K = kh+0..7, VGPR4-7 hold K = 16+kh+0..7.
// From f16 row-major memory that is exactly two contiguous 16B chunks per lane.
__device__ __forceinline__ v16h load_frag16(const _Float16* __restrict__ p, int lane) {
  int kh = (lane >> 4) << 3;
  v8h lo = *(const v8h*)(p + kh);        // K = kh .. kh+7       (global_load_b128)
  v8h hi = *(const v8h*)(p + 16 + kh);   // K = 16+kh .. 16+kh+7 (global_load_b128)
  v16h f;
#pragma unroll
  for (int e = 0; e < 8; ++e) { f[e] = lo[e]; f[e + 8] = hi[e]; }
  return f;
}

// f32 source variant with zero-padding (used only for edge_attr, K=16)
__device__ __forceinline__ int frag_k(int e, int kh) {
  int v = e >> 1, lo = e & 1;
  return ((v < 4) ? (kh + 2 * v) : (16 + kh + 2 * (v - 4))) + lo;
}
__device__ __forceinline__ v16h load_frag32(const float* __restrict__ p, int lane, int kvalid) {
  int kh = (lane >> 4) << 3;
  v16h f;
#pragma unroll
  for (int e = 0; e < 16; ++e) {
    int k = frag_k(e, kh);
    f[e] = (_Float16)((k < kvalid) ? p[k] : 0.0f);
  }
  return f;
}

// ---- one-time operand packing ----
__global__ void cvt_x16_k(const float* __restrict__ x, _Float16* __restrict__ x16) {
  size_t i = (size_t)blockIdx.x * blockDim.x + threadIdx.x;
  if (i < (size_t)NN * ND) x16[i] = (_Float16)x[i];
}

// weights -> f16, transposed to [col][K] so B fragments are contiguous in K
__global__ void pack_w_k(const float* __restrict__ node_w, const float* __restrict__ conv_w,
                         const float* __restrict__ edge_w, _Float16* __restrict__ nwT,
                         _Float16* __restrict__ cwT, _Float16* __restrict__ ewT) {
  int i = blockIdx.x * blockDim.x + threadIdx.x;
  if (i < HD * ND) {                       // node_wT [64][128]
    int col = i / ND, k = i % ND;
    nwT[i] = (_Float16)node_w[k * HD + col];
  }
  if (i < NL * HD * HD) {                  // conv_wT [3][64][64]
    int l = i / (HD * HD), r = i % (HD * HD);
    int col = r / HD, k = r % HD;
    cwT[i] = (_Float16)conv_w[(size_t)l * HD * HD + k * HD + col];
  }
  if (i < HD * EKP) {                      // edge_wT [64][32], zero-padded K
    int col = i / EKP, k = i % EKP;
    ewT[i] = (k < ED) ? (_Float16)edge_w[k * HD + col] : (_Float16)0.0f;
  }
}

// ---- node embedding: h = x16 @ node_wT^T + node_b (K=128), one wave / 16 rows ----
__global__ void node_gemm_wmma(const _Float16* __restrict__ A16, const _Float16* __restrict__ WT,
                               const float* __restrict__ bias, float* __restrict__ out) {
  int gw = (int)((blockIdx.x * blockDim.x + threadIdx.x) >> 5);
  int lane = threadIdx.x & 31;
  if (gw >= NN / 16) return;                 // wave-uniform; EXEC stays all-ones
  int row0 = gw << 4;
  int m = lane & 15;
  const _Float16* arow = A16 + (size_t)(row0 + m) * ND;
  v8f c[4] = {};
  for (int k0 = 0; k0 < ND; k0 += 32) {
    v16h a = load_frag16(arow + k0, lane);
#pragma unroll
    for (int nt = 0; nt < 4; ++nt) {
      v16h b = load_frag16(WT + (size_t)(nt * 16 + m) * ND + k0, lane);
      c[nt] = __builtin_amdgcn_wmma_f32_16x16x32_f16(false, a, false, b,
                                                     (short)0, c[nt], false, false);
    }
  }
  int rb = row0 + ((lane >> 4) << 3);        // C: lane holds col, VGPR v holds row rb+v
#pragma unroll
  for (int nt = 0; nt < 4; ++nt) {
    int col = nt * 16 + m;
    float bb = bias[col];
#pragma unroll
    for (int v = 0; v < 8; ++v)
      out[(size_t)(rb + v) * HD + col] = c[nt][v] + bb;
  }
}

// ---- conv GEMM with fused self-loop epilogue ----
// acc = dis^2 * (h16 @ W) + bias ;  xw16 = f16(h16 @ W)   (K=64)
__global__ void conv_gemm_wmma(const _Float16* __restrict__ A16, const _Float16* __restrict__ WT,
                               const float* __restrict__ bias, const float* __restrict__ dis,
                               float* __restrict__ acc, _Float16* __restrict__ xw16) {
  int gw = (int)((blockIdx.x * blockDim.x + threadIdx.x) >> 5);
  int lane = threadIdx.x & 31;
  if (gw >= NN / 16) return;
  int row0 = gw << 4;
  int m = lane & 15;
  const _Float16* arow = A16 + (size_t)(row0 + m) * HD;
  v8f c[4] = {};
#pragma unroll
  for (int k0 = 0; k0 < HD; k0 += 32) {
    v16h a = load_frag16(arow + k0, lane);
#pragma unroll
    for (int nt = 0; nt < 4; ++nt) {
      v16h b = load_frag16(WT + (size_t)(nt * 16 + m) * HD + k0, lane);
      c[nt] = __builtin_amdgcn_wmma_f32_16x16x32_f16(false, a, false, b,
                                                     (short)0, c[nt], false, false);
    }
  }
  int rb = row0 + ((lane >> 4) << 3);
#pragma unroll
  for (int nt = 0; nt < 4; ++nt) {
    int col = nt * 16 + m;
    float bb = bias[col];
#pragma unroll
    for (int v = 0; v < 8; ++v) {
      int rr = rb + v;
      float d = dis[rr];
      float val = c[nt][v];
      acc[(size_t)rr * HD + col] = d * d * val + bb;    // self-loop term
      xw16[(size_t)rr * HD + col] = (_Float16)val;      // gather source for scatter
    }
  }
}

// ---- edge embed: e = ea @ edge_w + edge_b, scattered to both endpoints ----
__global__ void edge_embed_wmma(const float* __restrict__ ea, const _Float16* __restrict__ ewT,
                                const float* __restrict__ bias, const int* __restrict__ rowI,
                                const int* __restrict__ colI, float* __restrict__ h) {
  int gw = (int)((blockIdx.x * blockDim.x + threadIdx.x) >> 5);
  int lane = threadIdx.x & 31;
  if (gw >= NE / 16) return;
  int e0 = gw << 4;
  int m = lane & 15;
  v16h a = load_frag32(ea + (size_t)(e0 + m) * ED, lane, ED);
  v8f c[4] = {};
#pragma unroll
  for (int nt = 0; nt < 4; ++nt) {
    v16h b = load_frag16(ewT + (size_t)(nt * 16 + m) * EKP, lane);
    c[nt] = __builtin_amdgcn_wmma_f32_16x16x32_f16(false, a, false, b,
                                                   (short)0, c[nt], false, false);
  }
  int mb = (lane >> 4) << 3;
#pragma unroll
  for (int v = 0; v < 8; ++v) {
    int ei = e0 + mb + v;
    int r = rowI[ei], cl = colI[ei];
#pragma unroll
    for (int nt = 0; nt < 4; ++nt) {
      int col = nt * 16 + m;
      float val = c[nt][v] + bias[col];
      atomic_add_f32(&h[(size_t)r * HD + col], val);
      atomic_add_f32(&h[(size_t)cl * HD + col], val);
    }
  }
}

// ---- degree / normalization ----
__global__ void deg_init_k(float* __restrict__ deg) {
  int n = blockIdx.x * blockDim.x + threadIdx.x;
  if (n < NN) deg[n] = 1.0f;  // self loop
}
__global__ void deg_edge_k(const int* __restrict__ colI, float* __restrict__ deg) {
  int e = blockIdx.x * blockDim.x + threadIdx.x;
  if (e < NE) atomic_add_f32(&deg[colI[e]], 1.0f);
}
__global__ void dis_k(const float* __restrict__ deg, float* __restrict__ dis) {
  int n = blockIdx.x * blockDim.x + threadIdx.x;
  if (n < NN) dis[n] = rsqrtf(deg[n]);
}

// h (f32, post edge-aggregation) -> h16 for first conv layer
__global__ void h16_cvt_k(const float* __restrict__ h, _Float16* __restrict__ h16) {
  size_t i = (size_t)blockIdx.x * blockDim.x + threadIdx.x;
  if (i < (size_t)NN * HD) h16[i] = (_Float16)h[i];
}

// acc[col] += dis[row]*dis[col]*xw[row]; 4 threads/edge, 16 cols each, f16 vector gathers
__global__ void conv_scatter_k(const _Float16* __restrict__ xw16, const int* __restrict__ rowI,
                               const int* __restrict__ colI, const float* __restrict__ dis,
                               float* __restrict__ acc) {
  size_t t = (size_t)blockIdx.x * blockDim.x + threadIdx.x;
  int e = (int)(t >> 2);
  if (e >= NE) return;
  int cseg = ((int)t & 3) << 4;
  int r = rowI[e], cl = colI[e];
  float coef = dis[r] * dis[cl];
  v8h a = *(const v8h*)(xw16 + (size_t)r * HD + cseg);       // 16B gather
  v8h b = *(const v8h*)(xw16 + (size_t)r * HD + cseg + 8);   // 16B gather
  float* d = acc + (size_t)cl * HD + cseg;
#pragma unroll
  for (int j = 0; j < 8; ++j) atomic_add_f32(d + j, coef * (float)a[j]);
#pragma unroll
  for (int j = 0; j < 8; ++j) atomic_add_f32(d + 8 + j, coef * (float)b[j]);
}

// h16 = relu(acc) as f16 (next layer input); hj gets the f32 relu for JK concat
__global__ void relu_copy_k(const float* __restrict__ acc, _Float16* __restrict__ h16,
                            float* __restrict__ hj, int layer) {
  size_t i = (size_t)blockIdx.x * blockDim.x + threadIdx.x;
  if (i >= (size_t)NN * HD) return;
  int n = (int)(i / HD), c = (int)(i % HD);
  float v = fmaxf(acc[i], 0.0f);
  h16[i] = (_Float16)v;
  hj[(size_t)n * JK + layer * HD + c] = v;
}

// ---- mean pool: batch is sorted -> one block per graph, no atomics ----
__device__ __forceinline__ int lower_bound_i(const int* __restrict__ a, int n, int key) {
  int lo = 0, hi = n;
  while (lo < hi) { int mid = (lo + hi) >> 1; if (a[mid] < key) lo = mid + 1; else hi = mid; }
  return lo;
}
__global__ void pool_graph_k(const float* __restrict__ hj, const int* __restrict__ batch,
                             float* __restrict__ pooled) {
  int g = blockIdx.x;
  int c = threadIdx.x;
  __shared__ int s_lo, s_hi;
  if (c == 0) {
    s_lo = lower_bound_i(batch, NN, g);
    s_hi = lower_bound_i(batch, NN, g + 1);
  }
  __syncthreads();
  if (c >= JK) return;
  int lo = s_lo, hi = s_hi;
  float s = 0.0f;
  for (int n = lo; n < hi; ++n) s += hj[(size_t)n * JK + c];
  int cnt = hi - lo;
  pooled[(size_t)g * JK + c] = s / (float)(cnt > 0 ? cnt : 1);
}

// ---- classifier (pooled already mean-ed) ----
__global__ void classifier_k(const float* __restrict__ pooled, const float* __restrict__ w1,
                             const float* __restrict__ b1, const float* __restrict__ w2,
                             const float* __restrict__ b2, float* __restrict__ out) {
  int g = blockIdx.x * blockDim.x + threadIdx.x;
  if (g >= NG) return;
  const float* p = pooled + (size_t)g * JK;
  float o = b2[0];
  for (int j = 0; j < HD / 2; ++j) {
    float s = b1[j];
    for (int k = 0; k < JK; ++k) s += p[k] * w1[k * (HD / 2) + j];
    o += fmaxf(s, 0.0f) * w2[j];
  }
  out[g] = o;
}

extern "C" void kernel_launch(void* const* d_in, const int* in_sizes, int n_in,
                              void* d_out, int out_size, void* d_ws, size_t ws_size,
                              hipStream_t stream) {
  const float* x      = (const float*)d_in[0];
  const float* ea     = (const float*)d_in[1];
  const float* node_w = (const float*)d_in[2];
  const float* node_b = (const float*)d_in[3];
  const float* edge_w = (const float*)d_in[4];
  const float* edge_b = (const float*)d_in[5];
  const float* conv_w = (const float*)d_in[6];
  const float* conv_b = (const float*)d_in[7];
  const float* cls_w1 = (const float*)d_in[8];
  const float* cls_b1 = (const float*)d_in[9];
  const float* cls_w2 = (const float*)d_in[10];
  const float* cls_b2 = (const float*)d_in[11];
  const int*   rowI   = (const int*)d_in[12];
  const int*   colI   = ((const int*)d_in[12]) + NE;
  const int*   batch  = (const int*)d_in[13];
  float*       out    = (float*)d_out;

  // workspace carve, 256B aligned regions (keeps 16B alignment for b128 loads)
  char* wp = (char*)d_ws;
  auto carve = [&](size_t bytes) {
    char* p = wp;
    wp += (bytes + 255) & ~(size_t)255;
    return p;
  };
  float*    h      = (float*)carve((size_t)NN * HD * 4);
  float*    acc    = (float*)carve((size_t)NN * HD * 4);
  float*    hj     = (float*)carve((size_t)NN * JK * 4);
  float*    deg    = (float*)carve((size_t)NN * 4);
  float*    dis    = (float*)carve((size_t)NN * 4);
  float*    pooled = (float*)carve((size_t)NG * JK * 4);
  _Float16* x16    = (_Float16*)carve((size_t)NN * ND * 2);
  _Float16* h16    = (_Float16*)carve((size_t)NN * HD * 2);
  _Float16* xw16   = (_Float16*)carve((size_t)NN * HD * 2);
  _Float16* nwT    = (_Float16*)carve((size_t)HD * ND * 2);
  _Float16* cwT    = (_Float16*)carve((size_t)NL * HD * HD * 2);
  _Float16* ewT    = (_Float16*)carve((size_t)HD * EKP * 2);

  const int T = 256;
  auto blk = [](size_t work, int t) { return (int)((work + t - 1) / t); };

  // 0) operand packing (tiny)
  pack_w_k<<<blk(NL * HD * HD, T), T, 0, stream>>>(node_w, conv_w, edge_w, nwT, cwT, ewT);
  cvt_x16_k<<<blk((size_t)NN * ND, T), T, 0, stream>>>(x, x16);

  // 1) node embedding (WMMA, K=128)
  node_gemm_wmma<<<blk((size_t)(NN / 16) * 32, T), T, 0, stream>>>(x16, nwT, node_b, h);

  // 2) degree + edge embedding scatter
  deg_init_k<<<blk(NN, T), T, 0, stream>>>(deg);
  edge_embed_wmma<<<blk((size_t)(NE / 16) * 32, T), T, 0, stream>>>(ea, ewT, edge_b,
                                                                    rowI, colI, h);
  deg_edge_k<<<blk(NE, T), T, 0, stream>>>(colI, deg);
  dis_k<<<blk(NN, T), T, 0, stream>>>(deg, dis);
  h16_cvt_k<<<blk((size_t)NN * HD, T), T, 0, stream>>>(h, h16);

  // 3) three GCN layers (GEMM fuses self-loop epilogue; scatter gathers f16)
  for (int i = 0; i < NL; ++i) {
    conv_gemm_wmma<<<blk((size_t)(NN / 16) * 32, T), T, 0, stream>>>(
        h16, cwT + (size_t)i * HD * HD, conv_b + i * HD, dis, acc, xw16);
    conv_scatter_k<<<blk((size_t)NE * 4, T), T, 0, stream>>>(xw16, rowI, colI, dis, acc);
    relu_copy_k<<<blk((size_t)NN * HD, T), T, 0, stream>>>(acc, h16, hj, i);
  }

  // 4) mean pool without atomics (batch sorted; one block per graph)
  pool_graph_k<<<NG, JK, 0, stream>>>(hj, batch, pooled);

  // 5) classifier
  classifier_k<<<blk(NG, T), T, 0, stream>>>(pooled, cls_w1, cls_b1, cls_w2, cls_b2, out);
}
